// DecoderLayer_33079838114732
// MI455X (gfx1250) — matrix operand order repaired and checked by reading, not compiled, not run on hardware
//
#include <hip/hip_runtime.h>
#include <hip/hip_bf16.h>
#include <math.h>

// ---------------------------------------------------------------------------
// Types for CDNA5 WMMA (wave32, 16x16x32 bf16 -> f32)
// ---------------------------------------------------------------------------
typedef __attribute__((ext_vector_type(16))) __bf16 bf16x16;
typedef __attribute__((ext_vector_type(8)))  __bf16 bf16x8;
typedef __attribute__((ext_vector_type(8)))  float  f32x8;

#define D_MODEL 512
#define N_HEAD  8
#define D_HEAD  64
#define SEQ     512
#define BATCH   16
#define DFF     2048
#define NTOK    (BATCH*SEQ)

#define OP_NONE 0
#define OP_RELU 1
#define OP_GELU 2

__device__ __forceinline__ float apply_op(float v, int op) {
  if (op == OP_RELU) return fmaxf(v, 0.f);
  if (op == OP_GELU) return 0.5f * v * (1.f + erff(v * 0.70710678118654752f));
  return v;
}

// ---------------------------------------------------------------------------
// CDNA5 async global->LDS copy (ASYNCcnt path, CDNA5 ISA).
// Each enabled lane copies 16 bytes: LDS[vdst] = MEM[vaddr].
// Generic 'shared' pointers carry the wave-relative LDS byte offset in the
// low 32 bits (aperture translation), which is exactly what VDST needs.
// ---------------------------------------------------------------------------
__device__ __forceinline__ void async_cp_b128(const __bf16* g, __bf16* l) {
  uint32_t lo = (uint32_t)(size_t)l;
  asm volatile("global_load_async_to_lds_b128 %0, %1, off"
               :: "v"(lo), "v"(g) : "memory");
}
__device__ __forceinline__ void wait_async() {
  asm volatile("s_wait_asynccnt 0x0" ::: "memory");
}

// ---------------------------------------------------------------------------
// Fragment loads from LDS; A/B use this file's contiguous k order (same for both), C/D as in CDNA5 ISA.
// A (16x32 bf16): lane L = row M=L%16; half=L/16:
//   elems e -> K = 16*half + e (contiguous)
// B (32x16 bf16) staged transposed in LDS as [n][k] (row stride 32):
//   lane L = col N=L%16; half=L/16: elems e -> K = 16*half + e (contiguous)
// C/D (16x16 f32): VGPR r, lane L -> M = r + 8*(L/16), N = L%16
// ---------------------------------------------------------------------------
__device__ __forceinline__ bf16x16 frag_a(const __bf16* As, int m0, int lane) {
  int m = m0 + (lane & 15);
  int half = lane >> 4;
  const __bf16* p = As + m * 32 + 16 * half;
  bf16x8 lo = *(const bf16x8*)p;
  bf16x8 hi = *(const bf16x8*)(p + 8);
  bf16x16 a;
#pragma unroll
  for (int i = 0; i < 8; ++i) { a[i] = lo[i]; a[i + 8] = hi[i]; }
  return a;
}

__device__ __forceinline__ bf16x16 frag_b(const __bf16* Bs, int n0, int lane) {
  int n = n0 + (lane & 15);
  int half = lane >> 4;
  const __bf16* p = Bs + n * 32 + 16 * half;
  bf16x8 lo = *(const bf16x8*)p;
  bf16x8 hi = *(const bf16x8*)(p + 8);
  bf16x16 b;
#pragma unroll
  for (int i = 0; i < 8; ++i) { b[i] = lo[i]; b[i + 8] = hi[i]; }
  return b;
}

__device__ __forceinline__ f32x8 wmma_bf16(bf16x16 a, bf16x16 b, f32x8 c) {
  return __builtin_amdgcn_wmma_f32_16x16x32_bf16(false, a, false, b,
                                                 (short)0, c, false, false);
}

// ---------------------------------------------------------------------------
// f32 -> bf16 conversion (grid-stride)
// ---------------------------------------------------------------------------
__global__ void cvt_f32_bf16(const float* __restrict__ s, __bf16* __restrict__ d,
                             long n) {
  long i = (long)blockIdx.x * blockDim.x + threadIdx.x;
  long stride = (long)gridDim.x * blockDim.x;
  for (; i < n; i += stride) d[i] = (__bf16)s[i];
}

// ---------------------------------------------------------------------------
// Dense GEMM: C[M,N] = act(A[M,K] @ W[K,N] + bias[N]).  A,W bf16 row-major.
// 128x128 block tile, BK=32, 256 threads = 8 waves in a 4(M) x 2(N) grid;
// each wave owns 32x64 (2 A-frags x 4 B-frags = 8 WMMA / k-step).
// A tile staged via async global->LDS (b128), B tile transposed via VGPRs.
// Double-buffered LDS so tile k+1 streams behind the WMMAs of tile k.
// ---------------------------------------------------------------------------
template <int OUT_BF16>
__global__ __launch_bounds__(256) void gemm_kernel(
    const __bf16* __restrict__ A, const __bf16* __restrict__ W,
    const float* __restrict__ bias, void* __restrict__ Cout,
    int M, int N, int K, int op) {
  __shared__ __bf16 As[2][128 * 32];
  __shared__ __bf16 Bs[2][128 * 32]; // transposed [n][k]
  const int ntiles = N >> 7;
  const int bm = blockIdx.x / ntiles;
  const int bn = blockIdx.x % ntiles;
  const int tid = threadIdx.x;
  const int wave = tid >> 5, lane = tid & 31;
  const int wm = (wave & 3) << 5;   // 0,32,64,96
  const int wn = (wave >> 2) << 6;  // 0,64
  const int ar = tid >> 1, ac = (tid & 1) << 4;  // A: 128 rows x 2 chunks of 16
  const int br = tid >> 3, bc = (tid & 7) << 4;  // W: 32 rows x 8 chunks of 16
  const __bf16* Ag = A + (size_t)(bm * 128 + ar) * K + ac;
  const __bf16* Wg = W + (size_t)br * N + bn * 128 + bc;

  f32x8 acc0[4] = {}; f32x8 acc1[4] = {};

  auto stage = [&](int buf, int k0) {
    // A: contiguous 32B per thread -> 2 async b128
    const __bf16* ga = Ag + k0;
    __bf16* la = &As[buf][ar * 32 + ac];
    async_cp_b128(ga, la);
    async_cp_b128(ga + 8, la + 8);
    // W: 16 contiguous cols per thread, scatter-transpose into [n][k]
    const __bf16* gw = Wg + (size_t)k0 * N;
    bf16x8 w0 = *(const bf16x8*)gw;
    bf16x8 w1 = *(const bf16x8*)(gw + 8);
#pragma unroll
    for (int i = 0; i < 8; ++i) {
      Bs[buf][(bc + i) * 32 + br] = w0[i];
      Bs[buf][(bc + 8 + i) * 32 + br] = w1[i];
    }
  };

  stage(0, 0);
  wait_async();
  __syncthreads();

  const int nsteps = K >> 5;
  for (int s = 0; s < nsteps; ++s) {
    const int cur = s & 1;
    if (s + 1 < nsteps) stage(cur ^ 1, (s + 1) << 5);
    bf16x16 a0 = frag_a(As[cur], wm, lane);
    bf16x16 a1 = frag_a(As[cur], wm + 16, lane);
#pragma unroll
    for (int j = 0; j < 4; ++j) {
      bf16x16 bj = frag_b(Bs[cur], wn + 16 * j, lane);
      acc0[j] = wmma_bf16(a0, bj, acc0[j]);
      acc1[j] = wmma_bf16(a1, bj, acc1[j]);
    }
    if (s + 1 < nsteps) wait_async();
    __syncthreads();
  }

  const int lm = (lane >> 4) << 3;
  const int lcol = lane & 15;
#pragma unroll
  for (int mi = 0; mi < 2; ++mi) {
    const int row0 = bm * 128 + wm + mi * 16 + lm;
#pragma unroll
    for (int j = 0; j < 4; ++j) {
      const int col = bn * 128 + wn + j * 16 + lcol;
      const float bi = bias[col];
      const f32x8 acc = mi ? acc1[j] : acc0[j];
#pragma unroll
      for (int r = 0; r < 8; ++r) {
        float v = apply_op(acc[r] + bi, op);
        size_t o = (size_t)(row0 + r) * N + col;
        if (OUT_BF16) ((__bf16*)Cout)[o] = (__bf16)v;
        else          ((float*)Cout)[o] = v;
      }
    }
  }
}

// ---------------------------------------------------------------------------
// Attention logits: out[b,h,q,k] = (Qh . Kh) * scale - 1e9*mask
// 64x64 tile, Kdim = 64 (2 k-steps). K rows are already [n][k] layout, so
// both stagings are contiguous -> async global->LDS.
// mask_mode 0: causal mask[q*Sk+k]; mode 1: padding mask[b*Sk+k].
// ---------------------------------------------------------------------------
__global__ __launch_bounds__(256) void attn_scores_kernel(
    const __bf16* __restrict__ Q, const __bf16* __restrict__ Kt,
    const float* __restrict__ mask, float* __restrict__ out,
    int Sq, int Sk, int mask_mode, float scale) {
  __shared__ __bf16 As[64 * 32];
  __shared__ __bf16 Bs[64 * 32];
  int idx = blockIdx.x;
  const int ntiles = Sk >> 6;
  const int mtiles = Sq >> 6;
  const int nt = idx % ntiles; idx /= ntiles;
  const int mt = idx % mtiles; idx /= mtiles;
  const int h = idx % N_HEAD;  const int b = idx / N_HEAD;
  const __bf16* Qb = Q  + (size_t)b * Sq * D_MODEL + h * D_HEAD;
  const __bf16* Kb = Kt + (size_t)b * Sk * D_MODEL + h * D_HEAD;
  float* ob = out + (size_t)(b * N_HEAD + h) * Sq * Sk;
  const int tid = threadIdx.x;
  const int wave = tid >> 5, lane = tid & 31;
  const int wm = (wave & 3) << 4;
  const int wn = (wave >> 2) << 5;
  const int ar = tid >> 2, ac = (tid & 3) << 3;
  f32x8 c0 = {}; f32x8 c1 = {};
#pragma unroll
  for (int k0 = 0; k0 < D_HEAD; k0 += 32) {
    async_cp_b128(Qb + (size_t)(mt * 64 + ar) * D_MODEL + k0 + ac,
                  &As[ar * 32 + ac]);
    async_cp_b128(Kb + (size_t)(nt * 64 + ar) * D_MODEL + k0 + ac,
                  &Bs[ar * 32 + ac]);
    wait_async();
    __syncthreads();
    bf16x16 a  = frag_a(As, wm, lane);
    bf16x16 b0 = frag_b(Bs, wn, lane);
    bf16x16 b1 = frag_b(Bs, wn + 16, lane);
    c0 = wmma_bf16(a, b0, c0);
    c1 = wmma_bf16(a, b1, c1);
    __syncthreads();
  }
  const int q0   = mt * 64 + wm + ((lane >> 4) << 3);
  const int col0 = nt * 64 + wn + (lane & 15);
  const int col1 = col0 + 16;
#pragma unroll
  for (int r = 0; r < 8; ++r) {
    int q = q0 + r;
    float m0 = (mask_mode == 0) ? mask[(size_t)q * Sk + col0]
                                : mask[(size_t)b * Sk + col0];
    float m1 = (mask_mode == 0) ? mask[(size_t)q * Sk + col1]
                                : mask[(size_t)b * Sk + col1];
    ob[(size_t)q * Sk + col0] = c0[r] * scale - 1e9f * m0;
    ob[(size_t)q * Sk + col1] = c1[r] * scale - 1e9f * m1;
  }
}

// ---------------------------------------------------------------------------
// Row softmax over Sk=512, in place (f32). One row per 128-thread block.
// ---------------------------------------------------------------------------
__global__ __launch_bounds__(128) void softmax_kernel(float* __restrict__ data) {
  const size_t row = blockIdx.x;
  float* p = data + row * 512;
  const int tid = threadIdx.x;
  float v[4];
  float mx = -3.0e38f;
#pragma unroll
  for (int i = 0; i < 4; ++i) { v[i] = p[tid + i * 128]; mx = fmaxf(mx, v[i]); }
  __shared__ float red[4];
#pragma unroll
  for (int o = 16; o > 0; o >>= 1) mx = fmaxf(mx, __shfl_xor(mx, o));
  if ((tid & 31) == 0) red[tid >> 5] = mx;
  __syncthreads();
  mx = fmaxf(fmaxf(red[0], red[1]), fmaxf(red[2], red[3]));
  float s = 0.f;
#pragma unroll
  for (int i = 0; i < 4; ++i) { v[i] = expf(v[i] - mx); s += v[i]; }
#pragma unroll
  for (int o = 16; o > 0; o >>= 1) s += __shfl_xor(s, o);
  __syncthreads();
  if ((tid & 31) == 0) red[tid >> 5] = s;
  __syncthreads();
  s = red[0] + red[1] + red[2] + red[3];
  const float inv = 1.f / s;
#pragma unroll
  for (int i = 0; i < 4; ++i) p[tid + i * 128] = v[i] * inv;
}

// ---------------------------------------------------------------------------
// ctx[b,q,h*64+n] = sum_k P[b,h,q,k] * V[b,k,h*64+n]; output bf16.
// P staged f32->bf16 into LDS; V staged transposed [n][k].
// ---------------------------------------------------------------------------
__global__ __launch_bounds__(256) void attn_av_kernel(
    const float* __restrict__ P, const __bf16* __restrict__ V,
    __bf16* __restrict__ ctx, int Sq, int Sk) {
  __shared__ __bf16 As[64 * 32];
  __shared__ __bf16 Bs[64 * 32];
  int idx = blockIdx.x;
  const int mtiles = Sq >> 6;
  const int mt = idx % mtiles; idx /= mtiles;
  const int h = idx % N_HEAD;  const int b = idx / N_HEAD;
  const float*  Pb = P + (size_t)(b * N_HEAD + h) * Sq * Sk;
  const __bf16* Vb = V + (size_t)b * Sk * D_MODEL + h * D_HEAD;
  __bf16* cb = ctx + (size_t)b * Sq * D_MODEL + h * D_HEAD;
  const int tid = threadIdx.x;
  const int wave = tid >> 5, lane = tid & 31;
  const int wm = (wave & 3) << 4;
  const int wn = (wave >> 2) << 5;
  const int ar = tid >> 2, ac = (tid & 3) << 3;
  const int br = tid >> 3, bc = (tid & 7) << 3;
  f32x8 c0 = {}; f32x8 c1 = {};
  for (int k0 = 0; k0 < Sk; k0 += 32) {
    const float* pr = Pb + (size_t)(mt * 64 + ar) * Sk + k0 + ac;
#pragma unroll
    for (int i = 0; i < 8; ++i) As[ar * 32 + ac + i] = (__bf16)pr[i];
    bf16x8 vv = *(const bf16x8*)(Vb + (size_t)(k0 + br) * D_MODEL + bc);
#pragma unroll
    for (int i = 0; i < 8; ++i) Bs[(bc + i) * 32 + br] = vv[i];
    __syncthreads();
    bf16x16 a  = frag_a(As, wm, lane);
    bf16x16 b0 = frag_b(Bs, wn, lane);
    bf16x16 b1 = frag_b(Bs, wn + 16, lane);
    c0 = wmma_bf16(a, b0, c0);
    c1 = wmma_bf16(a, b1, c1);
    __syncthreads();
  }
  const int q0 = mt * 64 + wm + ((lane >> 4) << 3);
  const int n0 = wn + (lane & 15);
#pragma unroll
  for (int r = 0; r < 8; ++r) {
    cb[(size_t)(q0 + r) * D_MODEL + n0]      = (__bf16)c0[r];
    cb[(size_t)(q0 + r) * D_MODEL + n0 + 16] = (__bf16)c1[r];
  }
}

// ---------------------------------------------------------------------------
// out = LayerNorm(a + b) * g + beta, row length D_MODEL=512.
// Also optionally emits a bf16 shadow copy (next GEMM's A operand).
// ---------------------------------------------------------------------------
__global__ __launch_bounds__(128) void add_ln_kernel(
    const float* __restrict__ a, const float* __restrict__ b,
    const float* __restrict__ g, const float* __restrict__ beta,
    float* __restrict__ out, __bf16* __restrict__ out_bf) {
  const size_t row = blockIdx.x;
  const float* pa = a + row * D_MODEL;
  const float* pb = b + row * D_MODEL;
  float* po = out + row * D_MODEL;
  const int tid = threadIdx.x;
  float v[4]; float s = 0.f;
#pragma unroll
  for (int i = 0; i < 4; ++i) { int c = tid + i * 128; v[i] = pa[c] + pb[c]; s += v[i]; }
  __shared__ float red[4];
#pragma unroll
  for (int o = 16; o > 0; o >>= 1) s += __shfl_xor(s, o);
  if ((tid & 31) == 0) red[tid >> 5] = s;
  __syncthreads();
  s = red[0] + red[1] + red[2] + red[3];
  const float mu = s * (1.f / D_MODEL);
  float var = 0.f;
#pragma unroll
  for (int i = 0; i < 4; ++i) { float d = v[i] - mu; var += d * d; }
#pragma unroll
  for (int o = 16; o > 0; o >>= 1) var += __shfl_xor(var, o);
  __syncthreads();
  if ((tid & 31) == 0) red[tid >> 5] = var;
  __syncthreads();
  var = (red[0] + red[1] + red[2] + red[3]) * (1.f / D_MODEL);
  const float rs = rsqrtf(var + 1e-6f);
#pragma unroll
  for (int i = 0; i < 4; ++i) {
    int c = tid + i * 128;
    float y = (v[i] - mu) * rs * g[c] + beta[c];
    po[c] = y;
    if (out_bf) out_bf[row * D_MODEL + c] = (__bf16)y;
  }
}

// ---------------------------------------------------------------------------
// Host orchestration
// ---------------------------------------------------------------------------
extern "C" void kernel_launch(void* const* d_in, const int* in_sizes, int n_in,
                              void* d_out, int out_size, void* d_ws, size_t ws_size,
                              hipStream_t stream) {
  (void)in_sizes; (void)n_in; (void)out_size; (void)ws_size;
  const float* x        = (const float*)d_in[0];
  const float* enc_fc   = (const float*)d_in[1];
  const float* enc_ff   = (const float*)d_in[2];
  const float* la_mask  = (const float*)d_in[3];
  const float* pad_mask = (const float*)d_in[4];
  const float* wW[8]; const float* wB[8];
  for (int i = 0; i < 8; ++i) { wW[i] = (const float*)d_in[5 + 2 * i]; wB[i] = (const float*)d_in[6 + 2 * i]; }
  // wW: 0=m1_wq 1=m1_wk 2=m1_wv 3=m1_do 4=m2_wq 5=m2_wk 6=m2_wv 7=m2_do
  const float* ffn_w1 = (const float*)d_in[21]; const float* ffn_b1 = (const float*)d_in[22];
  const float* ffn_w2 = (const float*)d_in[23]; const float* ffn_b2 = (const float*)d_in[24];
  const float* ln_g[3] = { (const float*)d_in[25], (const float*)d_in[27], (const float*)d_in[29] };
  const float* ln_b[3] = { (const float*)d_in[26], (const float*)d_in[28], (const float*)d_in[30] };

  // Output regions: out3 | aw1 | aw2
  float* out3 = (float*)d_out;
  float* aw1  = out3 + (size_t)BATCH * SEQ * D_MODEL;
  float* aw2  = aw1 + (size_t)BATCH * N_HEAD * SEQ * SEQ;

  // Workspace carve-out
  char* ws = (char*)d_ws;
  size_t off = 0;
  auto alloc_bf = [&](size_t elems) { __bf16* p = (__bf16*)(ws + off); off += elems * sizeof(__bf16); return p; };
  auto alloc_f  = [&](size_t elems) { float*  p = (float*)(ws + off);  off += elems * sizeof(float);  return p; };
  __bf16* wbf[8];
  for (int i = 0; i < 8; ++i) wbf[i] = alloc_bf((size_t)D_MODEL * D_MODEL);
  __bf16* w1bf  = alloc_bf((size_t)D_MODEL * DFF);
  __bf16* w2bf  = alloc_bf((size_t)DFF * D_MODEL);
  __bf16* xbf   = alloc_bf((size_t)NTOK * D_MODEL);
  __bf16* fcbf  = alloc_bf((size_t)NTOK * D_MODEL);
  __bf16* ffbf  = alloc_bf((size_t)NTOK * D_MODEL);
  __bf16* qbf   = alloc_bf((size_t)NTOK * D_MODEL);
  __bf16* kbf   = alloc_bf((size_t)NTOK * D_MODEL);
  __bf16* vbf   = alloc_bf((size_t)NTOK * D_MODEL);
  __bf16* ctxbf = alloc_bf((size_t)NTOK * D_MODEL);
  __bf16* obf   = alloc_bf((size_t)NTOK * D_MODEL);
  __bf16* midbf = alloc_bf((size_t)NTOK * DFF);
  float* proj = alloc_f((size_t)NTOK * D_MODEL);
  float* out1 = alloc_f((size_t)NTOK * D_MODEL);
  float* out2 = alloc_f((size_t)NTOK * D_MODEL);

  auto cvt = [&](const float* s, __bf16* d, size_t n) {
    cvt_f32_bf16<<<1024, 256, 0, stream>>>(s, d, (long)n);
  };
  auto gemm_f32 = [&](const __bf16* A, const __bf16* W, const float* bias,
                      float* C, int M, int N, int K, int op) {
    gemm_kernel<0><<<(M / 128) * (N / 128), 256, 0, stream>>>(A, W, bias, (void*)C, M, N, K, op);
  };
  auto gemm_bf = [&](const __bf16* A, const __bf16* W, const float* bias,
                     __bf16* C, int M, int N, int K, int op) {
    gemm_kernel<1><<<(M / 128) * (N / 128), 256, 0, stream>>>(A, W, bias, (void*)C, M, N, K, op);
  };

  // --- convert weights + inputs to bf16
  for (int i = 0; i < 8; ++i) cvt(wW[i], wbf[i], (size_t)D_MODEL * D_MODEL);
  cvt(ffn_w1, w1bf, (size_t)D_MODEL * DFF);
  cvt(ffn_w2, w2bf, (size_t)DFF * D_MODEL);
  cvt(x, xbf, (size_t)NTOK * D_MODEL);
  cvt(enc_fc, fcbf, (size_t)NTOK * D_MODEL);
  cvt(enc_ff, ffbf, (size_t)NTOK * D_MODEL);

  const float scale = 0.125f; // 1/sqrt(64)
  const int soft_rows = BATCH * N_HEAD * SEQ;
  const int attn_blocks = BATCH * N_HEAD * (SEQ / 64) * (SEQ / 64);
  const int av_blocks = BATCH * N_HEAD * (SEQ / 64);

  // --- MHA 1 (self, causal)
  gemm_bf(xbf, wbf[0], wB[0], qbf, NTOK, D_MODEL, D_MODEL, OP_NONE);
  gemm_bf(xbf, wbf[1], wB[1], kbf, NTOK, D_MODEL, D_MODEL, OP_NONE);
  gemm_bf(xbf, wbf[2], wB[2], vbf, NTOK, D_MODEL, D_MODEL, OP_NONE);
  attn_scores_kernel<<<attn_blocks, 256, 0, stream>>>(qbf, kbf, la_mask, aw1, SEQ, SEQ, 0, scale);
  softmax_kernel<<<soft_rows, 128, 0, stream>>>(aw1);
  attn_av_kernel<<<av_blocks, 256, 0, stream>>>(aw1, vbf, ctxbf, SEQ, SEQ);
  gemm_f32(ctxbf, wbf[3], wB[3], proj, NTOK, D_MODEL, D_MODEL, OP_GELU);
  add_ln_kernel<<<NTOK, 128, 0, stream>>>(proj, x, ln_g[0], ln_b[0], out1, obf);

  // --- MHA 2 (cross, padding mask)
  gemm_bf(obf,  wbf[4], wB[4], qbf, NTOK, D_MODEL, D_MODEL, OP_NONE);
  gemm_bf(fcbf, wbf[5], wB[5], kbf, NTOK, D_MODEL, D_MODEL, OP_NONE);
  gemm_bf(ffbf, wbf[6], wB[6], vbf, NTOK, D_MODEL, D_MODEL, OP_NONE);
  attn_scores_kernel<<<attn_blocks, 256, 0, stream>>>(qbf, kbf, pad_mask, aw2, SEQ, SEQ, 1, scale);
  softmax_kernel<<<soft_rows, 128, 0, stream>>>(aw2);
  attn_av_kernel<<<av_blocks, 256, 0, stream>>>(aw2, vbf, ctxbf, SEQ, SEQ);
  gemm_f32(ctxbf, wbf[7], wB[7], proj, NTOK, D_MODEL, D_MODEL, OP_GELU);
  add_ln_kernel<<<NTOK, 128, 0, stream>>>(proj, out1, ln_g[1], ln_b[1], out2, obf);

  // --- FFN
  gemm_bf(obf, w1bf, ffn_b1, midbf, NTOK, DFF, D_MODEL, OP_RELU);
  gemm_f32(midbf, w2bf, ffn_b2, proj, NTOK, D_MODEL, DFF, OP_NONE);
  add_ln_kernel<<<NTOK, 128, 0, stream>>>(proj, out2, ln_g[2], ln_b[2], out3, nullptr);
}
